// GRUD_26628797236075
// MI455X (gfx1250) — matrix operand order repaired and checked by reading, not compiled, not run on hardware
//
#include <hip/hip_runtime.h>
#include <hip/hip_bf16.h>

// ---------------- types ----------------
typedef _Float16 h16;
typedef __attribute__((ext_vector_type(16))) _Float16 v16h;
typedef __attribute__((ext_vector_type(8)))  _Float16 v8h;
typedef __attribute__((ext_vector_type(8)))  float    v8f;

// ---------------- problem dims ----------------
constexpr int Bc = 256, Tc = 1024, Fc = 128, Hc = 512, Oc = 64;
constexpr int M2 = Tc * Bc;        // 262144 time-major rows
constexpr int H3 = 3 * Hc;         // 1536

// ---------------- WMMA helpers ----------------
__device__ __forceinline__ v8f wmma16(v16h a, v16h b, v8f c) {
  return __builtin_amdgcn_wmma_f32_16x16x32_f16(false, a, false, b, (short)0, c,
                                                false, false);
}

// A fragment (16x32 f16): lane holds row M = lane&15; two contiguous 8-half
// chunks at K offsets (lane>>4)*8 and 16+(lane>>4)*8.
__device__ __forceinline__ v16h load_a(const h16* __restrict__ p, int ld,
                                       int m0, int kc) {
  const int lane = threadIdx.x & 31;
  const int row  = m0 + (lane & 15);
  const int kb   = kc + ((lane >> 4) << 3);
  const h16* q = p + (size_t)row * ld + kb;
  v8h lo = *(const v8h*)q;
  v8h hi = *(const v8h*)(q + 16);
  v16h a;
#pragma unroll
  for (int i = 0; i < 8; ++i) { a[i] = lo[i]; a[i + 8] = hi[i]; }
  return a;
}

// B fragment (32x16 f16) from global: lane holds column N = lane&15; K rows
// (lane>>4)*8+i and 16+(lane>>4)*8+i, i=0..7.
__device__ __forceinline__ v16h load_b(const h16* __restrict__ p, int ld,
                                       int kc, int n0) {
  const int lane = threadIdx.x & 31;
  const h16* q = p + (size_t)(kc + ((lane >> 4) << 3)) * ld + n0 + (lane & 15);
  v16h b;
#pragma unroll
  for (int i = 0; i < 8; ++i) { b[i] = q[(size_t)i * ld]; b[i + 8] = q[(size_t)(i + 16) * ld]; }
  return b;
}

// B fragment from an LDS-resident stripe laid out [3][512][16].
__device__ __forceinline__ v16h load_b_lds(const h16* sU, int mat, int kc) {
  const int lane = threadIdx.x & 31;
  const h16* q = sU + (((size_t)mat * Hc + kc + ((lane >> 4) << 3)) << 4) + (lane & 15);
  v16h b;
#pragma unroll
  for (int i = 0; i < 8; ++i) { b[i] = q[i << 4]; b[i + 8] = q[(i + 16) << 4]; }
  return b;
}

// ---------------- grid barrier (persistent scan) ----------------
__device__ __forceinline__ void gridbar(unsigned* cnt, volatile unsigned* gen) {
  __threadfence();
  __syncthreads();
  if (threadIdx.x == 0) {
    unsigned g = *gen;
    if (atomicAdd(cnt, 1u) == gridDim.x - 1) {
      *cnt = 0u;
      __threadfence();
      atomicAdd((unsigned*)gen, 1u);
    } else {
      while (*gen == g) __builtin_amdgcn_s_sleep(1);
    }
    __threadfence();
  }
  __syncthreads();
}

// ---------------- kernels ----------------
__global__ void k_init(unsigned* bar) {
  if (threadIdx.x < 128) bar[threadIdx.x] = 0u;
}

// Convert/pack weights to f16: U_z,U_r,U ; Wcat=[W*;V*] [256,1536]; W_gamma_h.
__global__ void k_pack(h16* uz, h16* ur, h16* uu, h16* wcat, h16* wg16,
                       const float* Uz, const float* Ur, const float* Uu,
                       const float* Wz, const float* Wr, const float* Ww,
                       const float* Vz, const float* Vr, const float* Vv,
                       const float* Wgh) {
  int idx = blockIdx.x * 256 + threadIdx.x;
  const int NU = Hc * Hc;  // 262144
  if (idx < 3 * NU) {
    int which = idx / NU, i = idx % NU;
    const float* s = (which == 0) ? Uz : (which == 1) ? Ur : Uu;
    h16* d = (which == 0) ? uz : (which == 1) ? ur : uu;
    d[i] = (h16)s[i];
    return;
  }
  idx -= 3 * NU;
  if (idx < 256 * H3) {
    int k = idx / H3, n = idx % H3;
    int sel = n >> 9, nn = n & 511;
    const float* Wm = (sel == 0) ? Wz : (sel == 1) ? Wr : Ww;
    const float* Vm = (sel == 0) ? Vz : (sel == 1) ? Vr : Vv;
    float v = (k < Fc) ? Wm[k * Hc + nn] : Vm[(k - Fc) * Hc + nn];
    wcat[idx] = (h16)v;
    return;
  }
  idx -= 256 * H3;
  if (idx < Fc * Hc) wg16[idx] = (h16)Wgh[idx];
}

// Elementwise: gamma_x decay + imputation; build time-major f16 operands:
// Acat[row][0:128]=x_eff, Acat[row][128:256]=1-m ; m16[row][f]=m. row = t*B+b.
__global__ void k_phaseA(const float* __restrict__ x,
                         const float* __restrict__ delta,
                         const float* __restrict__ m,
                         const float* __restrict__ xf,
                         const float* __restrict__ Wgx,
                         const float* __restrict__ bgx,
                         h16* __restrict__ acat, h16* __restrict__ m16) {
  int idx = blockIdx.x * 256 + threadIdx.x;   // M2*F threads
  int row = idx >> 7;
  int f = idx & 127;
  int t = row >> 8;        // B = 256
  int b = row & 255;
  size_t src = (size_t)b * Tc * Fc + (size_t)t * Fc + f;
  float mm = m[src];
  float gx = expf(-fmaxf(delta[src] * Wgx[f] + bgx[f], 0.f));
  float xr = gx * xf[src] + (1.f - gx) * 0.001f;
  float xe = (mm > 0.5f) ? xr : x[src];
  acat[(size_t)row * 256 + f]       = (h16)xe;
  acat[(size_t)row * 256 + 128 + f] = (h16)(1.f - mm);
  m16[(size_t)row * 128 + f]        = (h16)mm;
}

__device__ __forceinline__ void store_pre(float* __restrict__ pre, v8f acc,
                                          int rb, int cb,
                                          const float* bz, const float* br,
                                          const float* bw) {
  const int lane = threadIdx.x & 31;
  const int lm = (lane >> 4) << 3, ln = lane & 15;
  const int N = cb + ln;
  float bias = (N < Hc) ? bz[N] : (N < 2 * Hc) ? br[N - Hc] : bw[N - 2 * Hc];
#pragma unroll
  for (int v = 0; v < 8; ++v)
    pre[(size_t)(rb + lm + v) * H3 + N] = acc[v] + bias;
}

// pre = Acat[M2,256] @ Wcat[256,1536] + bias  (f16 WMMA, f32 accum)
// WG = 8 waves (2x4); each wave a 32x32 tile. grid (M2/64, H3/128).
__global__ void __launch_bounds__(256) k_gemm_pre(
    const h16* __restrict__ acat, const h16* __restrict__ wcat,
    const float* __restrict__ bz, const float* __restrict__ br,
    const float* __restrict__ bw, float* __restrict__ pre) {
  const int wid = threadIdx.x >> 5;
  const int rb = blockIdx.x * 64 + (wid & 1) * 32;
  const int cb = blockIdx.y * 128 + (wid >> 1) * 32;
  v8f a00 = {}, a01 = {}, a10 = {}, a11 = {};
  for (int kc = 0; kc < 256; kc += 32) {
    v16h fa0 = load_a(acat, 256, rb, kc);
    v16h fa1 = load_a(acat, 256, rb + 16, kc);
    v16h fb0 = load_b(wcat, H3, kc, cb);
    v16h fb1 = load_b(wcat, H3, kc, cb + 16);
    a00 = wmma16(fa0, fb0, a00);
    a01 = wmma16(fa0, fb1, a01);
    a10 = wmma16(fa1, fb0, a10);
    a11 = wmma16(fa1, fb1, a11);
  }
  store_pre(pre, a00, rb, cb, bz, br, bw);
  store_pre(pre, a01, rb, cb + 16, bz, br, bw);
  store_pre(pre, a10, rb + 16, cb, bz, br, bw);
  store_pre(pre, a11, rb + 16, cb + 16, bz, br, bw);
}

__device__ __forceinline__ void store_gam(h16* __restrict__ gam, v8f acc,
                                          int rb, int cb,
                                          const float* __restrict__ bgh) {
  const int lane = threadIdx.x & 31;
  const int lm = (lane >> 4) << 3, ln = lane & 15;
  const int N = cb + ln;
  const float bias = bgh[N];
#pragma unroll
  for (int v = 0; v < 8; ++v)
    gam[(size_t)(rb + lm + v) * Hc + N] =
        (h16)expf(-fmaxf(acc[v] + bias, 0.f));
}

// gamma_h = exp(-relu(m16[M2,128] @ Wg16[128,512] + b_gamma_h)) -> f16
__global__ void __launch_bounds__(256) k_gemm_gamma(
    const h16* __restrict__ m16, const h16* __restrict__ wg16,
    const float* __restrict__ bgh, h16* __restrict__ gam) {
  const int wid = threadIdx.x >> 5;
  const int rb = blockIdx.x * 64 + (wid & 1) * 32;
  const int cb = blockIdx.y * 128 + (wid >> 1) * 32;
  v8f a00 = {}, a01 = {}, a10 = {}, a11 = {};
  for (int kc = 0; kc < 128; kc += 32) {
    v16h fa0 = load_a(m16, 128, rb, kc);
    v16h fa1 = load_a(m16, 128, rb + 16, kc);
    v16h fb0 = load_b(wg16, Hc, kc, cb);
    v16h fb1 = load_b(wg16, Hc, kc, cb + 16);
    a00 = wmma16(fa0, fb0, a00);
    a01 = wmma16(fa0, fb1, a01);
    a10 = wmma16(fa1, fb0, a10);
    a11 = wmma16(fa1, fb1, a11);
  }
  store_gam(gam, a00, rb, cb, bgh);
  store_gam(gam, a01, rb, cb + 16, bgh);
  store_gam(gam, a10, rb + 16, cb, bgh);
  store_gam(gam, a11, rb + 16, cb + 16, bgh);
}

// Persistent cooperative scan: 64 WGs x 8 waves = 512 waves; wave w owns the
// 16x16 tile (m0=(w&15)*16 rows of B, n0=(w>>4)*16 cols of H) for all T steps,
// carrying h in registers. All 8 waves of a WG share n0, so the three U
// stripes U*[:, n0:n0+16] (3 x 16KB) are staged once into LDS and every
// B-fragment for all 1024 steps is a ds_load. 2 grid barriers per step.
__global__ void __launch_bounds__(256) k_scan(
    h16* __restrict__ hd16, h16* __restrict__ hdr16,
    const h16* __restrict__ gam, const float* __restrict__ pre,
    const h16* __restrict__ uz, const h16* __restrict__ ur,
    const h16* __restrict__ uu, const int* __restrict__ bsz,
    float* __restrict__ hfin, unsigned* __restrict__ bar) {
  __shared__ __align__(32) h16 sU[3 * Hc * 16];  // 48 KB

  const int wid = threadIdx.x >> 5;
  const int m0 = (((blockIdx.x & 1) << 3) + wid) << 4;  // (w & 15) * 16
  const int n0 = (blockIdx.x >> 1) << 4;                // uniform per WG
  const int lane = threadIdx.x & 31;
  const int lm = (lane >> 4) << 3;
  const int ln = lane & 15;
  int bRow[8];
#pragma unroll
  for (int v = 0; v < 8; ++v) bRow[v] = m0 + lm + v;
  const int nc = n0 + ln;

  // cooperative stage of U stripes into LDS: 3*512 rows of 16 halves (32B)
  for (int seg = threadIdx.x; seg < 3 * Hc; seg += 256) {
    const int mat = seg >> 9;   // 0:U_z 1:U_r 2:U
    const int k = seg & (Hc - 1);
    const h16* src =
        ((mat == 0) ? uz : (mat == 1) ? ur : uu) + (size_t)k * Hc + n0;
    *(v16h*)(sU + ((size_t)seg << 4)) = *(const v16h*)src;
  }
  __syncthreads();

  unsigned* cnt = bar;
  volatile unsigned* gen = bar + 64;

  v8f hcur = {};  // h0 = 0
  for (int t = 0; t < Tc; ++t) {
    const int bsv = bsz[t];
    const size_t rowb = (size_t)t * Bc;

    // prefetch next step's pre/gamma tiles (one lane per 16-col group;
    // columns of a group share cachelines) -> global_prefetch_b8
    if (t + 1 < Tc && ln == 0) {
      const size_t nrow = (size_t)(t + 1) * Bc;
#pragma unroll
      for (int v = 0; v < 8; ++v) {
        const size_t prow = (nrow + bRow[v]) * H3 + nc;
        __builtin_prefetch(&pre[prow], 0, 3);
        __builtin_prefetch(&pre[prow + Hc], 0, 3);
        __builtin_prefetch(&pre[prow + 2 * Hc], 0, 3);
        __builtin_prefetch(&gam[(nrow + bRow[v]) * Hc + nc], 0, 3);
      }
    }

    // h_d = mask ? gamma_h * h : h ; publish f16 copy for the GEMMs
    v8f hd;
#pragma unroll
    for (int v = 0; v < 8; ++v) {
      float g = (float)gam[(rowb + bRow[v]) * Hc + nc];
      hd[v] = (bRow[v] < bsv) ? g * hcur[v] : hcur[v];
      hd16[(size_t)bRow[v] * Hc + nc] = (h16)hd[v];
    }
    gridbar(cnt, gen);

    // z,r tiles: h_d @ U_z , h_d @ U_r   (K = 512, B-fragments from LDS)
    v8f accZ = {}, accR = {};
    for (int kc = 0; kc < Hc; kc += 32) {
      v16h fa = load_a(hd16, Hc, m0, kc);
      v16h fz = load_b_lds(sU, 0, kc);
      v16h fr = load_b_lds(sU, 1, kc);
      accZ = wmma16(fa, fz, accZ);
      accR = wmma16(fa, fr, accR);
    }
    v8f zt;
#pragma unroll
    for (int v = 0; v < 8; ++v) {
      size_t prow = (rowb + bRow[v]) * H3;
      float z = 1.f / (1.f + expf(-(accZ[v] + pre[prow + nc])));
      float r = 1.f / (1.f + expf(-(accR[v] + pre[prow + Hc + nc])));
      zt[v] = z;
      hdr16[(size_t)bRow[v] * Hc + nc] = (h16)(hd[v] * r);
    }
    gridbar(cnt, gen);

    // h_tilde tile: (h_d * r) @ U
    v8f accH = {};
    for (int kc = 0; kc < Hc; kc += 32) {
      v16h fa = load_a(hdr16, Hc, m0, kc);
      v16h fh = load_b_lds(sU, 2, kc);
      accH = wmma16(fa, fh, accH);
    }
#pragma unroll
    for (int v = 0; v < 8; ++v) {
      size_t prow = (rowb + bRow[v]) * H3;
      float ht = tanhf(accH[v] + pre[prow + 2 * Hc + nc]);
      hcur[v] =
          (bRow[v] < bsv) ? (1.f - zt[v]) * hd[v] + zt[v] * ht : hd[v];
    }
  }
#pragma unroll
  for (int v = 0; v < 8; ++v)
    hfin[(size_t)bRow[v] * Hc + nc] = hcur[v];
}

// BatchNorm (eval) + decoder + log_softmax. One block per batch row.
__global__ void __launch_bounds__(64) k_head(const float* __restrict__ hfin,
                                             const float* __restrict__ bnw,
                                             const float* __restrict__ bnb,
                                             const float* __restrict__ decW,
                                             const float* __restrict__ decb,
                                             float* __restrict__ out) {
  __shared__ float hbn[Hc];
  __shared__ float red[Oc];
  const int b = blockIdx.x, tid = threadIdx.x;
  const float rs = 0.9999950000374997f;  // 1/sqrt(1+1e-5)
  for (int h = tid; h < Hc; h += Oc)
    hbn[h] = hfin[b * Hc + h] * (bnw[h] * rs) + bnb[h];
  __syncthreads();
  float logit = decb[tid];
  for (int h = 0; h < Hc; ++h) logit += hbn[h] * decW[h * Oc + tid];
  red[tid] = logit;
  __syncthreads();
  for (int s = 32; s; s >>= 1) {
    if (tid < s) red[tid] = fmaxf(red[tid], red[tid + s]);
    __syncthreads();
  }
  float mx = red[0];
  __syncthreads();
  red[tid] = expf(logit - mx);
  __syncthreads();
  for (int s = 32; s; s >>= 1) {
    if (tid < s) red[tid] += red[tid + s];
    __syncthreads();
  }
  float lse = mx + logf(red[0]);
  out[b * Oc + tid] = logit - lse;                       // output [B,O]
  for (int h = tid; h < Hc; h += Oc)
    out[Bc * Oc + b * Hc + h] = hbn[h];                  // h_bn [B,H]
}

// ---------------- host launch ----------------
extern "C" void kernel_launch(void* const* d_in, const int* in_sizes, int n_in,
                              void* d_out, int out_size, void* d_ws,
                              size_t ws_size, hipStream_t stream) {
  const float* x     = (const float*)d_in[0];
  const float* delta = (const float*)d_in[1];
  const float* m     = (const float*)d_in[2];
  const float* xf    = (const float*)d_in[3];
  const int*   bsz   = (const int*)d_in[4];
  const float* W_r = (const float*)d_in[5];
  const float* U_r = (const float*)d_in[6];
  const float* V_r = (const float*)d_in[7];
  const float* b_r = (const float*)d_in[8];
  const float* W_z = (const float*)d_in[9];
  const float* U_z = (const float*)d_in[10];
  const float* V_z = (const float*)d_in[11];
  const float* b_z = (const float*)d_in[12];
  const float* Ww  = (const float*)d_in[13];
  const float* Uu  = (const float*)d_in[14];
  const float* Vv  = (const float*)d_in[15];
  const float* bb  = (const float*)d_in[16];
  const float* Wgx = (const float*)d_in[17];
  const float* bgx = (const float*)d_in[18];
  const float* Wgh = (const float*)d_in[19];
  const float* bgh = (const float*)d_in[20];
  const float* decW = (const float*)d_in[21];
  const float* decb = (const float*)d_in[22];
  const float* bnw  = (const float*)d_in[23];
  const float* bnb  = (const float*)d_in[24];
  float* out = (float*)d_out;

  // workspace carve-up (256B aligned)
  char* ws = (char*)d_ws;
  size_t off = 0;
  auto carve = [&](size_t bytes) {
    size_t o = off;
    off = (off + bytes + 255) & ~(size_t)255;
    return o;
  };
  h16*   acat  = (h16*)(ws + carve((size_t)M2 * 256 * sizeof(h16)));
  h16*   m16   = (h16*)(ws + carve((size_t)M2 * 128 * sizeof(h16)));
  float* pre   = (float*)(ws + carve((size_t)M2 * H3 * sizeof(float)));
  h16*   gam   = (h16*)(ws + carve((size_t)M2 * Hc * sizeof(h16)));
  h16*   uz16  = (h16*)(ws + carve((size_t)Hc * Hc * sizeof(h16)));
  h16*   ur16  = (h16*)(ws + carve((size_t)Hc * Hc * sizeof(h16)));
  h16*   uu16  = (h16*)(ws + carve((size_t)Hc * Hc * sizeof(h16)));
  h16*   wcat  = (h16*)(ws + carve((size_t)256 * H3 * sizeof(h16)));
  h16*   wg16  = (h16*)(ws + carve((size_t)Fc * Hc * sizeof(h16)));
  h16*   hd16  = (h16*)(ws + carve((size_t)Bc * Hc * sizeof(h16)));
  h16*   hdr16 = (h16*)(ws + carve((size_t)Bc * Hc * sizeof(h16)));
  float* hfin  = (float*)(ws + carve((size_t)Bc * Hc * sizeof(float)));
  unsigned* bar = (unsigned*)(ws + carve(256 * sizeof(unsigned)));
  (void)ws_size; (void)in_sizes; (void)n_in; (void)out_size;

  k_init<<<1, 128, 0, stream>>>(bar);
  {
    const int total = 3 * Hc * Hc + 256 * H3 + Fc * Hc;  // 1,245,184
    k_pack<<<(total + 255) / 256, 256, 0, stream>>>(
        uz16, ur16, uu16, wcat, wg16, U_z, U_r, Uu, W_z, W_r, Ww, V_z, V_r, Vv,
        Wgh);
  }
  k_phaseA<<<(M2 * Fc) / 256, 256, 0, stream>>>(x, delta, m, xf, Wgx, bgx,
                                                acat, m16);
  k_gemm_pre<<<dim3(M2 / 64, H3 / 128), 256, 0, stream>>>(acat, wcat, b_z, b_r,
                                                          bb, pre);
  k_gemm_gamma<<<dim3(M2 / 64, Hc / 128), 256, 0, stream>>>(m16, wg16, bgh,
                                                            gam);
  k_scan<<<64, 256, 0, stream>>>(hd16, hdr16, gam, pre, uz16, ur16, uu16, bsz,
                                 hfin, bar);
  k_head<<<Bc, Oc, 0, stream>>>(hfin, bnw, bnb, decW, decb, out);
}
